// Aligner_7722351199098
// MI455X (gfx1250) — compile-verified, matrix operand order checked by reading, not run on hardware
//
#include <hip/hip_runtime.h>
#include <hip/hip_fp16.h>

typedef __attribute__((ext_vector_type(16))) _Float16 v16h;
typedef __attribute__((ext_vector_type(8)))  _Float16 v8h;
typedef __attribute__((ext_vector_type(8)))  float    v8f;
typedef __attribute__((ext_vector_type(4)))  float    v4f;

namespace {
constexpr int   kB  = 16;
constexpr int   kC  = 512;
constexpr int   kTX = 1024;
constexpr int   kTY = 2048;
constexpr float kInvGamma = 1.0f / 10.0f;
constexpr int   LDSS = 72;   // half-stride: 144B rows, 16B-aligned frags, conflict-free b128
}

// ---------------------------------------------------------------------------
// Kernel 1: per-batch inclusive scan of durations -> Gaussian centers + mask bias
// ---------------------------------------------------------------------------
__global__ __launch_bounds__(1024)
void aligner_centers(const float* __restrict__ xlen, const float* __restrict__ xmask,
                     float* __restrict__ centers, float* __restrict__ bias) {
  __shared__ float s[kTX];
  const int b = blockIdx.x;
  const int t = threadIdx.x;
  const float v = xlen[b * kTX + t];
  s[t] = v;
  __syncthreads();
  for (int off = 1; off < kTX; off <<= 1) {
    float add = (t >= off) ? s[t - off] : 0.0f;
    __syncthreads();
    s[t] += add;
    __syncthreads();
  }
  centers[b * kTX + t] = s[t] - 0.5f * v;
  bias[b * kTX + t]    = -1e9f * (1.0f - xmask[b * kTX + t]);
}

// ---------------------------------------------------------------------------
// Kernel 2: per-(b,ty) softmax stats: max over tx and 1/sum(exp(l - max)).
// One wave32 per column; 8 columns per 256-thread block.
// ---------------------------------------------------------------------------
__global__ __launch_bounds__(256)
void aligner_colstats(const float* __restrict__ centers, const float* __restrict__ bias,
                      const int* __restrict__ y_offset,
                      float* __restrict__ colmax, float* __restrict__ colrcp) {
  const int wave = threadIdx.x >> 5;
  const int lane = threadIdx.x & 31;
  const int col  = blockIdx.x * 8 + wave;
  const int b    = col / kTY;
  const int ty   = col - b * kTY;
  const float pos = (float)(ty + y_offset[b]);
  const float* cb = centers + b * kTX;
  const float* bb = bias    + b * kTX;

  float mx = -3.0e38f;
#pragma unroll 4
  for (int k = lane; k < kTX; k += 32) {
    float d = cb[k] - pos;
    mx = fmaxf(mx, bb[k] - d * d * kInvGamma);
  }
#pragma unroll
  for (int off = 16; off > 0; off >>= 1) mx = fmaxf(mx, __shfl_xor(mx, off, 32));

  float sum = 0.0f;
#pragma unroll 4
  for (int k = lane; k < kTX; k += 32) {
    float d = cb[k] - pos;
    sum += __expf(bb[k] - d * d * kInvGamma - mx);
  }
#pragma unroll
  for (int off = 16; off > 0; off >>= 1) sum += __shfl_xor(sum, off, 32);

  if (lane == 0) {
    colmax[col] = mx;
    colrcp[col] = 1.0f / sum;
  }
}

// ---------------------------------------------------------------------------
// Kernel 3: fused softmax-weight x latent GEMM via WMMA f16 -> f32.
// Block = 256 threads (8 waves) computes a 64(C) x 64(TY) tile.
// K-step 64 (two WMMA K-chunks), double-buffered LDS, one barrier per step.
// Wave w: M-subtile = w&3; N-subtiles {2*(w>>2), 2*(w>>2)+1}; 4 WMMA/step.
// ---------------------------------------------------------------------------
__global__ __launch_bounds__(256)
void aligner_gemm(const float* __restrict__ xlat, const float* __restrict__ centers,
                  const float* __restrict__ bias, const float* __restrict__ colmax,
                  const float* __restrict__ colrcp, const int* __restrict__ y_offset,
                  float* __restrict__ out) {
  __shared__ _Float16 Xs[2][64 * LDSS];  // [m(c)][k]  f16 latents
  __shared__ _Float16 Ws[2][64 * LDSS];  // [n(ty)][k] f16 softmax weights

  const int n0 = blockIdx.x * 64;
  const int c0 = blockIdx.y * 64;
  const int b  = blockIdx.z;

  const int tid  = threadIdx.x;
  const int wave = tid >> 5;
  const int lane = tid & 31;
  const int m_sub  = wave & 3;
  const int n_base = (wave >> 2) * 2;

  // fragment addressing: lane l -> row l&15; K-half select by l>=16
  const int r   = lane & 15;
  const int khi = (lane >> 4) << 3;      // 0 or 8 (halves)

  // X staging map: thread -> (m = tid>>2, kq = (tid&3)*16), 16 floats/step
  const int sm  = tid >> 2;
  const int skq = (tid & 3) * 16;
  const float* xrow = xlat + ((size_t)b * kC + (c0 + sm)) * kTX + skq;
  // W staging map: thread -> (n = tid&63, kq = (tid>>6)*16), 16 weights/step
  const int wn  = tid & 63;
  const int wkq = (tid >> 6) * 16;
  const int ty  = n0 + wn;
  const float pos = (float)(ty + y_offset[b]);
  const float mx  = colmax[b * kTY + ty];
  const float rcp = colrcp[b * kTY + ty];
  const float* crow = centers + b * kTX + wkq;
  const float* brow = bias    + b * kTX + wkq;

  auto cvt_store_x = [&](int sel, v4f g0, v4f g1, v4f g2, v4f g3) {
    _Float16* d = &Xs[sel][sm * LDSS + skq];
    d[0]=(_Float16)g0.x; d[1]=(_Float16)g0.y; d[2]=(_Float16)g0.z; d[3]=(_Float16)g0.w;
    d[4]=(_Float16)g1.x; d[5]=(_Float16)g1.y; d[6]=(_Float16)g1.z; d[7]=(_Float16)g1.w;
    d[8]=(_Float16)g2.x; d[9]=(_Float16)g2.y; d[10]=(_Float16)g2.z; d[11]=(_Float16)g2.w;
    d[12]=(_Float16)g3.x; d[13]=(_Float16)g3.y; d[14]=(_Float16)g3.z; d[15]=(_Float16)g3.w;
  };
  auto cvt_store_w = [&](int sel, v4f c0v, v4f c1v, v4f c2v, v4f c3v,
                                   v4f b0v, v4f b1v, v4f b2v, v4f b3v) {
    _Float16* d = &Ws[sel][wn * LDSS + wkq];
    auto w1 = [&](float c, float bi) {
      float dd = c - pos;
      return (_Float16)(__expf(bi - dd * dd * kInvGamma - mx) * rcp);
    };
    d[0]=w1(c0v.x,b0v.x); d[1]=w1(c0v.y,b0v.y); d[2]=w1(c0v.z,b0v.z); d[3]=w1(c0v.w,b0v.w);
    d[4]=w1(c1v.x,b1v.x); d[5]=w1(c1v.y,b1v.y); d[6]=w1(c1v.z,b1v.z); d[7]=w1(c1v.w,b1v.w);
    d[8]=w1(c2v.x,b2v.x); d[9]=w1(c2v.y,b2v.y); d[10]=w1(c2v.z,b2v.z); d[11]=w1(c2v.w,b2v.w);
    d[12]=w1(c3v.x,b3v.x); d[13]=w1(c3v.y,b3v.y); d[14]=w1(c3v.z,b3v.z); d[15]=w1(c3v.w,b3v.w);
  };

  // prologue: stage chunk 0 into buffer 0
  {
    v4f g0 = *(const v4f*)(xrow + 0);
    v4f g1 = *(const v4f*)(xrow + 4);
    v4f g2 = *(const v4f*)(xrow + 8);
    v4f g3 = *(const v4f*)(xrow + 12);
    v4f cc0 = *(const v4f*)(crow + 0),  cc1 = *(const v4f*)(crow + 4);
    v4f cc2 = *(const v4f*)(crow + 8),  cc3 = *(const v4f*)(crow + 12);
    v4f bb0 = *(const v4f*)(brow + 0),  bb1 = *(const v4f*)(brow + 4);
    v4f bb2 = *(const v4f*)(brow + 8),  bb3 = *(const v4f*)(brow + 12);
    cvt_store_x(0, g0, g1, g2, g3);
    cvt_store_w(0, cc0, cc1, cc2, cc3, bb0, bb1, bb2, bb3);
  }
  __syncthreads();

  v8f acc0 = {};
  v8f acc1 = {};
  constexpr int kSteps = kTX / 64;

  for (int it = 0; it < kSteps; ++it) {
    const int cur = it & 1;
    const int nxt = cur ^ 1;
    const int k0n = (it + 1) * 64;
    const bool has_next = (it + 1) < kSteps;

    // 1) fragment loads from current buffer (12 x ds_load_b128)
    union Frag { v16h v; v8h h[2]; };
    Frag A0, A1, B00, B01, B10, B11;
    const _Float16* Xb  = &Xs[cur][(m_sub * 16 + r) * LDSS];
    const _Float16* Wb0 = &Ws[cur][(n_base * 16 + r) * LDSS];
    const _Float16* Wb1 = Wb0 + 16 * LDSS;
    A0.h[0]  = *(const v8h*)&Xb[khi];       A0.h[1]  = *(const v8h*)&Xb[16 + khi];
    A1.h[0]  = *(const v8h*)&Xb[32 + khi];  A1.h[1]  = *(const v8h*)&Xb[48 + khi];
    B00.h[0] = *(const v8h*)&Wb0[khi];      B00.h[1] = *(const v8h*)&Wb0[16 + khi];
    B01.h[0] = *(const v8h*)&Wb0[32 + khi]; B01.h[1] = *(const v8h*)&Wb0[48 + khi];
    B10.h[0] = *(const v8h*)&Wb1[khi];      B10.h[1] = *(const v8h*)&Wb1[16 + khi];
    B11.h[0] = *(const v8h*)&Wb1[32 + khi]; B11.h[1] = *(const v8h*)&Wb1[48 + khi];

    // 2) issue next chunk's global loads early (latency hidden under WMMAs)
    v4f g0{}, g1{}, g2{}, g3{};
    v4f cc0{}, cc1{}, cc2{}, cc3{}, bb0{}, bb1{}, bb2{}, bb3{};
    if (has_next) {
      g0 = *(const v4f*)(xrow + k0n + 0);
      g1 = *(const v4f*)(xrow + k0n + 4);
      g2 = *(const v4f*)(xrow + k0n + 8);
      g3 = *(const v4f*)(xrow + k0n + 12);
      cc0 = *(const v4f*)(crow + k0n + 0);  cc1 = *(const v4f*)(crow + k0n + 4);
      cc2 = *(const v4f*)(crow + k0n + 8);  cc3 = *(const v4f*)(crow + k0n + 12);
      bb0 = *(const v4f*)(brow + k0n + 0);  bb1 = *(const v4f*)(brow + k0n + 4);
      bb2 = *(const v4f*)(brow + k0n + 8);  bb3 = *(const v4f*)(brow + k0n + 12);
    }

    // 3) four WMMAs, alternating accumulators (independent back-to-back issue)
    acc0 = __builtin_amdgcn_wmma_f32_16x16x32_f16(false, A0.v, false, B00.v,
                                                  (short)0, acc0, false, false);
    acc1 = __builtin_amdgcn_wmma_f32_16x16x32_f16(false, A0.v, false, B10.v,
                                                  (short)0, acc1, false, false);
    acc0 = __builtin_amdgcn_wmma_f32_16x16x32_f16(false, A1.v, false, B01.v,
                                                  (short)0, acc0, false, false);
    acc1 = __builtin_amdgcn_wmma_f32_16x16x32_f16(false, A1.v, false, B11.v,
                                                  (short)0, acc1, false, false);

    // 4) convert + stage into the other buffer; 5) single barrier orders both
    //    hazards (our reads of `cur` vs others' writes next iter, and our
    //    writes of `nxt` vs others' reads next iter)
    if (has_next) {
      cvt_store_x(nxt, g0, g1, g2, g3);
      cvt_store_w(nxt, cc0, cc1, cc2, cc3, bb0, bb1, bb2, bb3);
    }
    __syncthreads();
  }

  // store: D layout VGPR i -> M = i + 8*(lane>=16); N = lane&15
  const int mbase = c0 + m_sub * 16 + ((lane >> 4) << 3);
  const int nb0   = n0 + n_base * 16 + (lane & 15);
  const int nb1   = nb0 + 16;
  float* ob = out + (size_t)b * kC * kTY;
#pragma unroll
  for (int i = 0; i < 8; ++i) {
    ob[(size_t)(mbase + i) * kTY + nb0] = acc0[i];
    ob[(size_t)(mbase + i) * kTY + nb1] = acc1[i];
  }
}

// ---------------------------------------------------------------------------
extern "C" void kernel_launch(void* const* d_in, const int* in_sizes, int n_in,
                              void* d_out, int out_size, void* d_ws, size_t ws_size,
                              hipStream_t stream) {
  (void)in_sizes; (void)n_in; (void)out_size; (void)ws_size;
  const float* x_latents = (const float*)d_in[0];
  const float* x_lengths = (const float*)d_in[1];
  const float* x_mask    = (const float*)d_in[2];
  /* d_in[3] = y_len: unused, TY is static */
  const int*   y_offset  = (const int*)d_in[4];
  float* out = (float*)d_out;

  char* ws = (char*)d_ws;   // 384 KB used
  float* centers = (float*)(ws);
  float* bias    = (float*)(ws + sizeof(float) * (size_t)kB * kTX);
  float* colmax  = (float*)(ws + sizeof(float) * (size_t)kB * kTX * 2);
  float* colrcp  = (float*)(ws + sizeof(float) * ((size_t)kB * kTX * 2 + (size_t)kB * kTY));

  aligner_centers<<<kB, kTX, 0, stream>>>(x_lengths, x_mask, centers, bias);
  aligner_colstats<<<(kB * kTY) / 8, 256, 0, stream>>>(centers, bias, y_offset,
                                                       colmax, colrcp);
  dim3 grid(kTY / 64, kC / 64, kB);
  aligner_gemm<<<grid, 256, 0, stream>>>(x_latents, centers, bias,
                                         colmax, colrcp, y_offset, out);
}
